// TSoftmaxLayer_63196148793812
// MI455X (gfx1250) — matrix operand order repaired and checked by reading, not compiled, not run on hardware
//
#include <hip/hip_runtime.h>

// TSoftmaxLayer: out[bt,j] = sum_i softmax_i(w[bt,i,j]) * x[bt,i]
// B*T = 16384 tiles of (I=128) x (J=128). HBM-bound: ~1.08 GB @ 23.3 TB/s ~ 46 us.
// One workgroup (256 thr / 8 waves) per tile:
//   P1: stage 64KB weight tile + x into LDS via CDNA5 async HBM->LDS DMA
//       (global_load_async_to_lds_b128, tracked by ASYNCcnt) - no VGPR bounce.
//   P2: fp32 softmax over i per column j (LDS-resident, bank-conflict free)
//   P3: combine via v_wmma_f32_16x16x32_f16, one 16-wide j-tile per wave

#if __has_builtin(__builtin_amdgcn_global_load_async_to_lds_b128)
#define USE_ASYNC_LDS 1
#else
#define USE_ASYNC_LDS 0
#endif

typedef __attribute__((ext_vector_type(16))) _Float16 v16h;
typedef __attribute__((ext_vector_type(8)))  float    v8f;
typedef __attribute__((ext_vector_type(4)))  float    v4f;
typedef __attribute__((ext_vector_type(4)))  int      v4i;

#if USE_ASYNC_LDS
typedef __attribute__((address_space(1))) v4i* gv4i_ptr;  // global (AS1) int4*
typedef __attribute__((address_space(3))) v4i* lv4i_ptr;  // LDS (AS3) int4*
#endif

#define I_DIM 128
#define J_DIM 128

__global__ __launch_bounds__(256) void tsoftmax_mix_kernel(
    const float* __restrict__ x_g,   // (BT, 128)
    const float* __restrict__ w_g,   // (BT, 128, 128)
    float*       __restrict__ out_g) // (BT, 128)
{
    extern __shared__ float smem[];
    float* w_s  = smem;                    // 128*128 floats: raw w, then exp(w-m)
    float* x_s  = smem + I_DIM * J_DIM;    // 128 floats
    float* rZ_s = x_s + I_DIM;             // 128 floats: 1/Z per column j

    const int       tid = threadIdx.x;
    const long long bt  = blockIdx.x;

    // ---------------- Phase 1: stage tile into LDS (single HBM pass) ----------
#if USE_ASYNC_LDS
    {
        const char* gsrc = (const char*)(w_g + bt * (long long)(I_DIM * J_DIM));
        char*       ldst = (char*)w_s;
        #pragma unroll
        for (int it = 0; it < (I_DIM * J_DIM * 4 / 16) / 256; ++it) {
            const int byte_off = (it * 256 + tid) * 16;      // coalesced b128
            __builtin_amdgcn_global_load_async_to_lds_b128(
                (gv4i_ptr)(gsrc + byte_off), (lv4i_ptr)(ldst + byte_off),
                /*offset=*/0, /*cpol=*/0);
        }
        if (tid < (I_DIM * 4) / 16) {                        // 32 lanes for x
            __builtin_amdgcn_global_load_async_to_lds_b128(
                (gv4i_ptr)((const char*)(x_g + bt * I_DIM) + tid * 16),
                (lv4i_ptr)((char*)x_s + tid * 16),
                /*offset=*/0, /*cpol=*/0);
        }
        // Drain this wave's ASYNCcnt, then barrier publishes LDS to the WG.
#if __has_builtin(__builtin_amdgcn_s_wait_asynccnt)
        __builtin_amdgcn_s_wait_asynccnt(0);
#else
        asm volatile("s_wait_asynccnt 0x0" ::: "memory");
#endif
    }
#else
    {
        const v4f* __restrict__ src = (const v4f*)(w_g + bt * (long long)(I_DIM * J_DIM));
        v4f* dst = (v4f*)w_s;
        #pragma unroll
        for (int it = 0; it < (I_DIM * J_DIM / 4) / 256; ++it) {
            const int idx = it * 256 + tid;                   // coalesced b128
            dst[idx] = __builtin_nontemporal_load(src + idx); // streamed once: NT
        }
        if (tid < I_DIM / 4) {
            ((v4f*)x_s)[tid] = ((const v4f*)(x_g + bt * I_DIM))[tid];
        }
    }
#endif
    __syncthreads();

    // ---------------- Phase 2: fp32 softmax over i, per column j --------------
    if (tid < J_DIM) {
        const int j = tid;  // lanes span j at fixed i -> distinct LDS banks
        float m = -3.402823466e38f;
        #pragma unroll 8
        for (int i = 0; i < I_DIM; ++i)
            m = fmaxf(m, w_s[i * J_DIM + j]);
        float Z = 0.0f;
        #pragma unroll 8
        for (int i = 0; i < I_DIM; ++i) {
            const float e = __expf(w_s[i * J_DIM + j] - m);  // TEMPERATURE == 1.0
            Z += e;
            w_s[i * J_DIM + j] = e;                          // overwrite in place
        }
        rZ_s[j] = 1.0f / Z;
    }
    __syncthreads();

    // ---------------- Phase 3: WMMA combine, one 16-col j-tile per wave -------
    // D = A x B + C, fp32 accum. A(16x32 f16) = x chunk replicated over all 16 M
    // rows; B(32x16 f16) = normalized probabilities. Every D row == out[j-tile].
    const int wv   = tid >> 5;          // wave id 0..7
    const int lane = tid & 31;
    const int j0   = wv * 16;
    const int N    = lane & 15;         // B/D column within tile
    const int g    = lane >> 4;         // half-wave group
    const float rZ = rZ_s[j0 + N];      // per-lane column scale

    v8f c = {};
    #pragma unroll
    for (int kc = 0; kc < I_DIM; kc += 32) {
        v16h a, b;
        // A layout (16-bit 16x32): lanes g==0 hold K = 0..15,
        //                          lanes g==1 hold K = 16..31 (matches B)
        #pragma unroll
        for (int h = 0; h < 8; ++h) {
            a[h]     = (_Float16)x_s[kc + g * 16 + h];
            a[h + 8] = (_Float16)x_s[kc + g * 16 + 8 + h];
        }
        // B layout (16-bit 32x16): lane group g holds K = g*16 + (0..15), N = lane&15
        #pragma unroll
        for (int h = 0; h < 16; ++h) {
            b[h] = (_Float16)(w_s[(kc + g * 16 + h) * J_DIM + j0 + N] * rZ);
        }
        c = __builtin_amdgcn_wmma_f32_16x16x32_f16(
                /*neg_a=*/false, a, /*neg_b=*/false, b,
                /*c_mod=*/(short)0, c, /*reuse_a=*/false, /*reuse_b=*/false);
    }

    // D layout: VGPR0 lanes 0-15 hold M=0 row == out[j0+N]; rows are identical.
    if (lane < 16) {
        __builtin_nontemporal_store(c[0], out_g + bt * (long long)J_DIM + j0 + N);
    }
}

extern "C" void kernel_launch(void* const* d_in, const int* in_sizes, int n_in,
                              void* d_out, int out_size, void* d_ws, size_t ws_size,
                              hipStream_t stream) {
    const float* x   = (const float*)d_in[0];  // (B,T,I) fp32
    const float* w   = (const float*)d_in[1];  // (B,T,I,J) fp32
    float*       out = (float*)d_out;          // (B,T,J) fp32

    const int BT = in_sizes[0] / I_DIM;        // 16384 tiles
    const size_t smem_bytes = (size_t)(I_DIM * J_DIM + 2 * I_DIM) * sizeof(float); // 65 KB

    tsoftmax_mix_kernel<<<dim3(BT), dim3(256), smem_bytes, stream>>>(x, w, out);
}
